// MeshGraphNet_1760936591507
// MI455X (gfx1250) — matrix operand order repaired and checked by reading, not compiled, not run on hardware
//
#include <hip/hip_runtime.h>
#include <cstdint>

typedef __bf16 bf16;
typedef __attribute__((ext_vector_type(16))) __bf16 v16bf;
typedef __attribute__((ext_vector_type(8)))  float  v8f;

enum { MODE_BF16 = 0, MODE_F32PAD = 1, MODE_CAT_EDGE = 2, MODE_CAT_NODE = 3 };
enum { ACT_NONE = 0, ACT_RELU = 1, ACT_MISH = 2 };

union Frag { v16bf v; uint32_t u[8]; };

struct GemmArgs {
  const bf16*  A;      // MODE_BF16 input, row stride = Kpad
  const float* Af32;   // MODE_F32PAD input, row stride = Ksrc
  const bf16*  h;      // latent node features [*,128]
  const bf16*  e;      // latent edge features [*,128]
  const float* agg;    // fp32 aggregation    [*,128]
  const int*   isrc;   // edge_index[0]
  const int*   idst;   // edge_index[1]
  const bf16*  Wt;     // transposed bf16 weights [128][Kpad]
  const float* bias;   // [128]
  bf16*        outB;   // [M][128]
  int M, Ksrc, Kpad;
};

__device__ __forceinline__ uint32_t pkbf(float a, float b) {
  union { bf16 h[2]; uint32_t u; } p;
  p.h[0] = (bf16)a; p.h[1] = (bf16)b;
  return p.u;
}

// Fetch one 16-col segment (16 bf16 = 8 dwords) of A row r into registers.
template <int MODE>
__device__ __forceinline__ void fetch_a8(uint32_t o[8], const GemmArgs& g, int r, int gk,
                                         int rDst, int rSrc) {
  if (r >= g.M) {
#pragma unroll
    for (int j = 0; j < 8; ++j) o[j] = 0u;
    return;
  }
  if (MODE == MODE_F32PAD) {
    // Ksrc is 16 or 8; segments are 16-wide, so only 3 cases (no per-element guards).
    const int nv = g.Ksrc - gk;
    const float* s = g.Af32 + (size_t)r * g.Ksrc + gk;
    if (nv >= 16) {
#pragma unroll
      for (int j = 0; j < 8; ++j) o[j] = pkbf(s[2 * j], s[2 * j + 1]);
    } else if (nv >= 8) {
#pragma unroll
      for (int j = 0; j < 4; ++j) o[j] = pkbf(s[2 * j], s[2 * j + 1]);
#pragma unroll
      for (int j = 4; j < 8; ++j) o[j] = 0u;
    } else {
#pragma unroll
      for (int j = 0; j < 8; ++j) o[j] = 0u;
    }
    return;
  }
  const bf16* s;
  if (MODE == MODE_BF16) {
    s = g.A + (size_t)r * g.Kpad + gk;
  } else if (MODE == MODE_CAT_EDGE) {
    if (gk < 128)      s = g.h + (size_t)rDst * 128 + gk;
    else if (gk < 256) s = g.h + (size_t)rSrc * 128 + (gk - 128);
    else               s = g.e + (size_t)r * 128 + (gk - 256);
  } else { // MODE_CAT_NODE
    if (gk < 128) {
      s = g.h + (size_t)r * 128 + gk;
    } else {
      const float* f = g.agg + (size_t)r * 128 + (gk - 128);
#pragma unroll
      for (int j = 0; j < 8; ++j) o[j] = pkbf(f[2 * j], f[2 * j + 1]);
      return;
    }
  }
  const uint32_t* su = (const uint32_t*)s;
#pragma unroll
  for (int j = 0; j < 8; ++j) o[j] = su[j];
}

// 128x128-tile GEMM: out[M][128] = act(A[M][Kpad] @ Wt^T + bias).
// bf16 WMMA, fp32 accum, double-buffered LDS, one barrier per K-chunk.
template <int MODE, int ACT>
__global__ __launch_bounds__(256) void gemm128(GemmArgs g) {
  __shared__ __align__(16) bf16 sA[2][128 * 34]; // 32 data + 2 pad bf16/row (17 dwords)
  __shared__ __align__(16) bf16 sB[2][128 * 34]; // Wt tile [n][k], same padding

  const int tid   = threadIdx.x;
  const int lane  = tid & 31;
  const int wave  = tid >> 5;
  const int mt    = wave & 3;  // 4 waves along M (32 rows each)
  const int nt    = wave >> 2; // 2 waves along N (64 cols each)
  const int mBase = blockIdx.x * 128;
  const int lr    = tid >> 1;        // tile row handled by this thread
  const int lh    = (tid & 1) << 4;  // 16-col segment within the K-chunk

  // Hoist gather indices out of the K-loop (CAT_EDGE only).
  int rDst = 0, rSrc = 0;
  if (MODE == MODE_CAT_EDGE) {
    const int rr = mBase + lr;
    if (rr < g.M) { rDst = g.idst[rr]; rSrc = g.isrc[rr]; }
  }

  v8f acc[2][4] = {};

  uint32_t pa[8], pb[8];
  // Prefetch chunk 0 into registers, stage into buffer 0.
  fetch_a8<MODE>(pa, g, mBase + lr, lh, rDst, rSrc);
  {
    const uint32_t* su = (const uint32_t*)(g.Wt + (size_t)lr * g.Kpad + lh);
#pragma unroll
    for (int j = 0; j < 8; ++j) pb[j] = su[j];
  }
  {
    uint32_t* da = (uint32_t*)&sA[0][lr * 34 + lh];
    uint32_t* db = (uint32_t*)&sB[0][lr * 34 + lh];
#pragma unroll
    for (int j = 0; j < 8; ++j) { da[j] = pa[j]; db[j] = pb[j]; }
  }

  const int nChunks = g.Kpad >> 5;
  for (int kc = 0; kc < nChunks; ++kc) {
    const int  cur  = kc & 1;
    const bool more = (kc + 1 < nChunks);
    if (more) {
      // Register prefetch of next chunk (global loads overlap this chunk's WMMAs).
      fetch_a8<MODE>(pa, g, mBase + lr, ((kc + 1) << 5) + lh, rDst, rSrc);
      const uint32_t* su =
          (const uint32_t*)(g.Wt + (size_t)lr * g.Kpad + ((kc + 1) << 5) + lh);
#pragma unroll
      for (int j = 0; j < 8; ++j) pb[j] = su[j];
      if (kc + 2 < nChunks)
        __builtin_prefetch(g.Wt + (size_t)lr * g.Kpad + ((kc + 2) << 5) + lh, 0, 3);
    }
    __syncthreads(); // buf[cur] stores visible; buf[cur^1] readers from kc-1 done

    const int half = lane >> 4;
    const int rl   = lane & 15;

    Frag a2[2];
#pragma unroll
    for (int am = 0; am < 2; ++am) {
      const uint32_t* rp = (const uint32_t*)&sA[cur][(mt * 32 + am * 16 + rl) * 34];
#pragma unroll
      for (int p = 0; p < 8; ++p) // ISA 16-bit A 16x32 layout
        a2[am].u[p] = rp[(p & 3) + half * 4 + ((p >> 2) << 3)];
    }
    Frag b4[4];
#pragma unroll
    for (int bn = 0; bn < 4; ++bn) {
      const uint32_t* cp = (const uint32_t*)&sB[cur][(nt * 64 + bn * 16 + rl) * 34];
#pragma unroll
      for (int v = 0; v < 8; ++v) // ISA 16-bit B 32x16 layout: same K order as A
        b4[bn].u[v] = cp[(v & 3) + half * 4 + ((v >> 2) << 3)];
    }
#pragma unroll
    for (int am = 0; am < 2; ++am)
#pragma unroll
      for (int bn = 0; bn < 4; ++bn)
        acc[am][bn] = __builtin_amdgcn_wmma_f32_16x16x32_bf16(
            false, a2[am].v, false, b4[bn].v, (short)0, acc[am][bn], false, false);

    if (more) {
      uint32_t* da = (uint32_t*)&sA[cur ^ 1][lr * 34 + lh];
      uint32_t* db = (uint32_t*)&sB[cur ^ 1][lr * 34 + lh];
#pragma unroll
      for (int j = 0; j < 8; ++j) { da[j] = pa[j]; db[j] = pb[j]; }
    }
  }

  const int half = lane >> 4;
  const int rl   = lane & 15;
#pragma unroll
  for (int am = 0; am < 2; ++am) {
#pragma unroll
    for (int bn = 0; bn < 4; ++bn) {
      const int   col = nt * 64 + bn * 16 + rl;
      const float bb  = g.bias[col];
#pragma unroll
      for (int v = 0; v < 8; ++v) {
        const int row = mBase + mt * 32 + am * 16 + half * 8 + v;
        if (row < g.M) {
          float x = acc[am][bn][v] + bb;
          if (ACT == ACT_RELU) {
            x = fmaxf(x, 0.0f);
          } else if (ACT == ACT_MISH) {
            float sp = (x > 20.0f) ? x : log1pf(expf(x));
            x = x * tanhf(sp);
          }
          g.outB[(size_t)row * 128 + col] = (bf16)x;
        }
      }
    }
  }
}

// fp32 [K][N] weights -> transposed bf16 [N][Kpad] (zero-padded K)
__global__ void conv_wt(const float* __restrict__ W, bf16* __restrict__ Wt,
                        int K, int N, int Kpad) {
  int i = blockIdx.x * blockDim.x + threadIdx.x;
  if (i >= N * Kpad) return;
  int n = i / Kpad, k = i - n * Kpad;
  Wt[i] = (k < K) ? (bf16)W[(size_t)k * N + n] : (bf16)0.0f;
}

__global__ void fillz(float* __restrict__ p, int n) {
  int i = blockIdx.x * blockDim.x + threadIdx.x;
  if (i < n) p[i] = 0.0f;
}

__global__ void scatter_add(const bf16* __restrict__ m, const int* __restrict__ srcIdx,
                            float* __restrict__ agg, int E) {
  int i = blockIdx.x * blockDim.x + threadIdx.x;
  if (i >= E * 128) return;
  int e2 = i >> 7, f = i & 127;
  atomicAdd(&agg[(size_t)srcIdx[e2] * 128 + f], (float)m[i]);
}

// Final decoder layer: [M,128] bf16 @ Wt[3][128] + bias -> fp32 out (no activation)
__global__ void dec_out(const bf16* __restrict__ X, const bf16* __restrict__ Wt,
                        const float* __restrict__ bias, float* __restrict__ out, int M) {
  int r = blockIdx.x * blockDim.x + threadIdx.x;
  if (r >= M) return;
  const bf16* xr = X + (size_t)r * 128;
#pragma unroll
  for (int o = 0; o < 3; ++o) {
    float s = bias[o];
    const bf16* w = Wt + o * 128;
#pragma unroll 8
    for (int k = 0; k < 128; ++k) s = fmaf((float)xr[k], (float)w[k], s);
    out[(size_t)r * 3 + o] = s;
  }
}

static void launch_gemm(int mode, int act, const GemmArgs& g, hipStream_t s) {
  dim3 grid((g.M + 127) / 128), blk(256);
  if (mode == MODE_F32PAD && act == ACT_MISH)      gemm128<MODE_F32PAD, ACT_MISH><<<grid, blk, 0, s>>>(g);
  else if (mode == MODE_BF16 && act == ACT_MISH)   gemm128<MODE_BF16,   ACT_MISH><<<grid, blk, 0, s>>>(g);
  else if (mode == MODE_BF16 && act == ACT_RELU)   gemm128<MODE_BF16,   ACT_RELU><<<grid, blk, 0, s>>>(g);
  else if (mode == MODE_BF16 && act == ACT_NONE)   gemm128<MODE_BF16,   ACT_NONE><<<grid, blk, 0, s>>>(g);
  else if (mode == MODE_CAT_EDGE)                  gemm128<MODE_CAT_EDGE, ACT_RELU><<<grid, blk, 0, s>>>(g);
  else if (mode == MODE_CAT_NODE)                  gemm128<MODE_CAT_NODE, ACT_RELU><<<grid, blk, 0, s>>>(g);
}

extern "C" void kernel_launch(void* const* d_in, const int* in_sizes, int n_in,
                              void* d_out, int out_size, void* d_ws, size_t ws_size,
                              hipStream_t stream) {
  const float* x     = (const float*)d_in[0];
  const int*   eidx  = (const int*)d_in[1];
  const float* eattr = (const float*)d_in[2];
  const int Nn = in_sizes[0] / 16;
  const int Ne = in_sizes[2] / 8;
  const int* srcIdx = eidx;       // edge_index[0] (x_j / scatter target)
  const int* dstIdx = eidx + Ne;  // edge_index[1] (x_i)

  // 48 Linear layers in params insertion order:
  // node_enc(0-2), edge_enc(3-5), node_dec(6-8), edge_dec(9-11),
  // procs p: edge_mlp(+0..2), node_fn(+3..5), edge_fn(+6..8), base 12+9p.
  int dK[48], dN[48], dKp[48];
  size_t wOff[48];
  {
    int t = 0;
    auto add3 = [&](int k0, int nl) {
      dK[t] = k0;  dN[t] = 128; t++;
      dK[t] = 128; dN[t] = 128; t++;
      dK[t] = 128; dN[t] = nl;  t++;
    };
    add3(16, 128); add3(8, 128); add3(128, 3); add3(128, 3);
    for (int p = 0; p < 4; ++p) { add3(384, 128); add3(256, 128); add3(128, 128); }
  }
  size_t off = 0;
  for (int i = 0; i < 48; ++i) {
    dKp[i] = ((dK[i] + 31) / 32) * 32;
    wOff[i] = off;
    size_t sz = (size_t)dN[i] * dKp[i];
    off += (sz + 127) & ~(size_t)127;
  }

  auto AL = [](size_t v) { return (v + 255) & ~(size_t)255; };
  char* base = (char*)d_ws;
  size_t pos = 0;
  bf16* WT = (bf16*)(base + pos); pos += AL(off * sizeof(bf16));
  bf16* H  = (bf16*)(base + pos); pos += AL((size_t)Nn * 128 * sizeof(bf16));
  bf16* Eb = (bf16*)(base + pos); pos += AL((size_t)Ne * 128 * sizeof(bf16));
  bf16* Mb = (bf16*)(base + pos); pos += AL((size_t)Ne * 128 * sizeof(bf16));
  bf16* T1 = (bf16*)(base + pos); pos += AL((size_t)Ne * 128 * sizeof(bf16));
  bf16* T2 = (bf16*)(base + pos); pos += AL((size_t)Ne * 128 * sizeof(bf16));
  float* AGG = (float*)(base + pos); pos += AL((size_t)Nn * 128 * sizeof(float));
  if (pos > ws_size) return;

  auto Wf = [&](int i) { return (const float*)d_in[3 + 2 * i]; };
  auto Bf = [&](int i) { return (const float*)d_in[3 + 2 * i + 1]; };

  for (int i = 0; i < 48; ++i) {
    int total = dN[i] * dKp[i];
    conv_wt<<<dim3((total + 255) / 256), dim3(256), 0, stream>>>(
        Wf(i), WT + wOff[i], dK[i], dN[i], dKp[i]);
  }

  auto gemm = [&](int mode, int act, int M, const bf16* A, const float* Af32,
                  int Ksrc, int li, bf16* out) {
    GemmArgs g{};
    g.A = A; g.Af32 = Af32; g.h = H; g.e = Eb; g.agg = AGG;
    g.isrc = srcIdx; g.idst = dstIdx;
    g.Wt = WT + wOff[li]; g.bias = Bf(li);
    g.outB = out; g.M = M; g.Ksrc = Ksrc; g.Kpad = dKp[li];
    launch_gemm(mode, act, g, stream);
  };

  // Encoders (Mish, none on last layer)
  gemm(MODE_F32PAD, ACT_MISH, Nn, nullptr, x, 16, 0, T1);
  gemm(MODE_BF16,   ACT_MISH, Nn, T1, nullptr, 0, 1, T2);
  gemm(MODE_BF16,   ACT_NONE, Nn, T2, nullptr, 0, 2, H);
  gemm(MODE_F32PAD, ACT_MISH, Ne, nullptr, eattr, 8, 3, T1);
  gemm(MODE_BF16,   ACT_MISH, Ne, T1, nullptr, 0, 4, T2);
  gemm(MODE_BF16,   ACT_NONE, Ne, T2, nullptr, 0, 5, Eb);

  // 4 processor blocks
  for (int p = 0; p < 4; ++p) {
    const int lb = 12 + p * 9;
    // edge_mlp on cat(h[dst], h[src], e)
    gemm(MODE_CAT_EDGE, ACT_RELU, Ne, nullptr, nullptr, 0, lb + 0, T1);
    gemm(MODE_BF16,     ACT_RELU, Ne, T1, nullptr, 0, lb + 1, T2);
    gemm(MODE_BF16,     ACT_NONE, Ne, T2, nullptr, 0, lb + 2, Mb);
    // scatter-add m onto edge_index[0]
    {
      int tot = Nn * 128;
      fillz<<<dim3((tot + 255) / 256), dim3(256), 0, stream>>>(AGG, tot);
      int st = Ne * 128;
      scatter_add<<<dim3((st + 255) / 256), dim3(256), 0, stream>>>(Mb, srcIdx, AGG, Ne);
    }
    // node_fn on cat(h, agg)
    gemm(MODE_CAT_NODE, ACT_RELU, Nn, nullptr, nullptr, 0, lb + 3, T1);
    gemm(MODE_BF16,     ACT_RELU, Nn, T1, nullptr, 0, lb + 4, T2);
    gemm(MODE_BF16,     ACT_NONE, Nn, T2, nullptr, 0, lb + 5, H);
    // edge_fn on m
    gemm(MODE_BF16, ACT_RELU, Ne, Mb, nullptr, 0, lb + 6, T1);
    gemm(MODE_BF16, ACT_RELU, Ne, T1, nullptr, 0, lb + 7, T2);
    gemm(MODE_BF16, ACT_NONE, Ne, T2, nullptr, 0, lb + 8, Eb);
  }

  // Decoders (Mish, Mish, linear N=3)
  float* outp = (float*)d_out;
  gemm(MODE_BF16, ACT_MISH, Nn, H,  nullptr, 0, 6, T1);
  gemm(MODE_BF16, ACT_MISH, Nn, T1, nullptr, 0, 7, T2);
  dec_out<<<dim3((Nn + 255) / 256), dim3(256), 0, stream>>>(
      T2, WT + wOff[8], Bf(8), outp, Nn);
  gemm(MODE_BF16, ACT_MISH, Ne, Eb, nullptr, 0, 9, T1);
  gemm(MODE_BF16, ACT_MISH, Ne, T1, nullptr, 0, 10, T2);
  dec_out<<<dim3((Ne + 255) / 256), dim3(256), 0, stream>>>(
      T2, WT + wOff[11], Bf(11), outp + (size_t)Nn * 3, Ne);
}
